// VectorQuantizer_77773267796003
// MI455X (gfx1250) — compile-verified, matrix operand order checked by reading, not run on hardware
//
#include <hip/hip_runtime.h>
#include <hip/hip_bf16.h>

// Problem constants (from reference)
#define K_CODES 512
#define DIMS    64
#define BB      32
#define TT      8192
#define NROWS   (BB * TT)          // 262144
#define NTILES  (NROWS / 16)       // 16384
#define MAIN_BLOCKS 512
#define TPB     256
#define TILES_PER_WAVE 4           // 512 blocks * 8 waves * 4 = 16384

typedef __attribute__((ext_vector_type(16))) __bf16 v16bf;
typedef __attribute__((ext_vector_type(8)))  float  v8f;

__device__ __forceinline__ unsigned short f2bf(float f) {
  union { float f; unsigned u; } v; v.f = f;
  // round-to-nearest-even f32 -> bf16
  return (unsigned short)((v.u + 0x7FFFu + ((v.u >> 16) & 1u)) >> 16);
}

__device__ __forceinline__ unsigned fbits(float f) {
  union { float f; unsigned u; } v; v.f = f; return v.u;
}
__device__ __forceinline__ float bitsf(unsigned u) {
  union { unsigned u; float f; } v; v.u = u; return v.f;
}
// monotone (order-preserving) float -> u32 map for unsigned compare
__device__ __forceinline__ unsigned ordenc(float f) {
  unsigned u = fbits(f);
  return (u & 0x80000000u) ? ~u : (u | 0x80000000u);
}
__device__ __forceinline__ float orddec(unsigned o) {
  return bitsf((o & 0x80000000u) ? (o & 0x7fffffffu) : ~o);
}

// ---------------------------------------------------------------------------
// Prep: swizzle codebook into WMMA B-fragment order (bf16) + |e|^2 in f32.
// B fragment layout for V_WMMA_F32_16X16X32_BF16 (32x16 B tile):
//   lane holds column N = lane%16; lanes 0-15: VGPR0-3 -> K 0..7, VGPR4-7 -> K 16..23
//                                  lanes 16-31: VGPR0-3 -> K 8..15, VGPR4-7 -> K 24..31
//   each VGPR packs (k_even, k_odd) bf16.
//   ushort idx = (((c*2 + h)*32) + lane)*16 + e,  c = code tile, h = k half (0/1)
// ---------------------------------------------------------------------------
__global__ __launch_bounds__(TPB) void vq_prep(const float* __restrict__ emb,
                                               unsigned short* __restrict__ wsB,
                                               float* __restrict__ wsEn) {
  int tid = blockIdx.x * blockDim.x + threadIdx.x;   // 32768 threads total
  int e    = tid & 15;
  int lane = (tid >> 4) & 31;
  int ch   = tid >> 9;
  int h    = ch & 1;
  int c    = ch >> 1;
  int p    = e >> 1;
  int odd  = e & 1;
  int k    = ((p < 4) ? 2 * p : 2 * p + 8) + ((lane >= 16) ? 8 : 0) + 32 * h + odd;
  int code = c * 16 + (lane & 15);
  wsB[tid] = f2bf(emb[code * DIMS + k]);
  if (tid < K_CODES) {
    float s = 0.f;
    for (int i = 0; i < DIMS; ++i) { float v = emb[tid * DIMS + i]; s = __builtin_fmaf(v, v, s); }
    wsEn[tid] = s;
  }
}

// ---------------------------------------------------------------------------
// Main: per wave, 16-row x tiles vs all 512 codes via bf16 WMMA.
// ---------------------------------------------------------------------------
__global__ __launch_bounds__(TPB) void vq_main(const float* __restrict__ xin,
                                               const float* __restrict__ emb,
                                               const unsigned short* __restrict__ wsB,
                                               const float* __restrict__ wsEn,
                                               float* __restrict__ outQ,
                                               float* __restrict__ outI,
                                               float* __restrict__ wsP) {
  __shared__ unsigned char sB[65536];   // full swizzled codebook in LDS
  __shared__ float sEn[K_CODES];        // |e|^2 in LDS (2 KB)
  {
    const uint4* s = (const uint4*)wsB;
    uint4*       d = (uint4*)sB;
    for (int i = threadIdx.x; i < 4096; i += TPB) d[i] = s[i];
    for (int i = threadIdx.x; i < K_CODES; i += TPB) sEn[i] = wsEn[i];
  }
  __syncthreads();

  const int lane = threadIdx.x & 31;
  const int w    = threadIdx.x >> 5;
  const int r    = lane & 15;                 // row-within-tile (t) / code-within-tile
  const int koff = (lane >= 16) ? 8 : 0;

  const unsigned char* bbase = sB + (size_t)lane * 32;
  const float*         ebase = sEn + r;

  float waveLoss = 0.f;

  for (int it = 0; it < TILES_PER_WAVE; ++it) {
    int tau  = (blockIdx.x * 8 + w) * TILES_PER_WAVE + it;
    int bIdx = tau >> 9;                      // T/16 = 512 tiles per batch
    int t0   = (tau & 511) << 4;
    const float* xb = xin + ((size_t)bIdx * DIMS) * TT + (size_t)(t0 + r);

    // ---- Build A fragments (16x64 bf16 as two 16x32 halves), f32 |x|^2 ----
    unsigned short afrag[2][16];
    float xn = 0.f;
#pragma unroll
    for (int h = 0; h < 2; ++h) {
#pragma unroll
      for (int p = 0; p < 8; ++p) {
        int   k  = ((p < 4) ? 2 * p : 2 * p + 8) + koff + 32 * h;
        float f0 = xb[(size_t)k * TT];
        float f1 = xb[(size_t)(k + 1) * TT];
        xn = __builtin_fmaf(f0, f0, xn);
        xn = __builtin_fmaf(f1, f1, xn);
        afrag[h][2 * p]     = f2bf(f0);
        afrag[h][2 * p + 1] = f2bf(f1);
      }
    }
    v16bf a0, a1;
    __builtin_memcpy(&a0, afrag[0], 32);
    __builtin_memcpy(&a1, afrag[1], 32);
    xn += __shfl_xor(xn, 16, 32);             // full |x|^2 for row r

    float best[8];
    int   bidx[8];
#pragma unroll
    for (int j = 0; j < 8; ++j) { best[j] = 3.4e38f; bidx[j] = 0; }

    // ---- 32 code tiles x 2 WMMAs: dist_cmp = |e|^2 - 2 * (x . e) ----
    // unroll-4: enough to pipeline ds_loads past WMMA waits without spilling
#pragma unroll 4
    for (int c = 0; c < 32; ++c) {
      v16bf b0 = *(const v16bf*)(bbase + (size_t)c * 2048);
      v16bf b1 = *(const v16bf*)(bbase + (size_t)c * 2048 + 1024);
      float en = ebase[c * 16];               // LDS, batched into same dscnt wait
      v8f acc = {};
      acc = __builtin_amdgcn_wmma_f32_16x16x32_bf16(false, a0, false, b0,
                                                    (short)0, acc, false, false);
      acc = __builtin_amdgcn_wmma_f32_16x16x32_bf16(false, a1, false, b1,
                                                    (short)0, acc, false, false);
      int code = c * 16 + r;
#pragma unroll
      for (int j = 0; j < 8; ++j) {
        float s = __builtin_fmaf(-2.0f, acc[j], en);
        // strict < keeps earliest (lowest) code on ties within this lane
        bool t = s < best[j];
        best[j] = t ? s : best[j];
        bidx[j] = t ? code : bidx[j];
      }
    }

    // ---- branchless argmin butterfly over the 16 lanes holding each row ----
    // lexicographic u64 key: (order-encoded distance << 32) | code index
    unsigned long long key[8];
#pragma unroll
    for (int j = 0; j < 8; ++j)
      key[j] = ((unsigned long long)ordenc(best[j]) << 32) | (unsigned)bidx[j];
#pragma unroll
    for (int s = 1; s < 16; s <<= 1) {
#pragma unroll
      for (int j = 0; j < 8; ++j) {
        unsigned long long ok = __shfl_xor(key[j], s, 32);
        key[j] = (ok < key[j]) ? ok : key[j];
      }
    }
    float bd[8];
    int   bi[8];
#pragma unroll
    for (int j = 0; j < 8; ++j) {
      bd[j] = orddec((unsigned)(key[j] >> 32));
      bi[j] = (int)(unsigned)(key[j] & 0xffffffffu);
    }

    // ---- loss:  ||e - x||^2 = |x|^2 + (|e|^2 - 2 x.e) = xn + bd ----
    float sumRows = 0.f;
    int   jb = (lane >= 16) ? 8 : 0;          // lower half holds rows 0-7, upper 8-15
#pragma unroll
    for (int j = 0; j < 8; ++j) {
      float xr = __shfl(xn, jb + j, 32);      // lane m holds row m's |x|^2
      sumRows += bd[j] + xr;
    }
    waveLoss += __shfl(sumRows, 0, 32) + __shfl(sumRows, 16, 32);

    // ---- distribute per-row winning index to the lane owning that t ----
    int ownedIdx = bi[0];                     // lane owns row (lane&7)+(lane>=16?8:0)
#pragma unroll
    for (int j = 1; j < 8; ++j) ownedIdx = ((lane & 7) == j) ? bi[j] : ownedIdx;
    int src   = (r < 8) ? r : (r + 8);
    int myIdx = __shfl(ownedIdx, src, 32);

    if (lane < 16) outI[(size_t)bIdx * TT + t0 + r] = (float)myIdx;

    // ---- quantized output: gather f32 codebook row, coalesced stores ----
    const float* erow = emb + (size_t)myIdx * DIMS;
    float* ob = outQ + ((size_t)bIdx * DIMS) * TT + (size_t)(t0 + r);
    int dh = lane >> 4;
#pragma unroll
    for (int d0 = 0; d0 < DIMS; d0 += 2) {
      int d = d0 + dh;
      ob[(size_t)d * TT] = erow[d];
    }
  }

  if (lane == 0) wsP[blockIdx.x * 8 + w] = waveLoss;
}

// ---------------------------------------------------------------------------
// Finalize: deterministic fixed-order reduction of 4096 wave partials.
// ---------------------------------------------------------------------------
__global__ __launch_bounds__(TPB) void vq_final(const float* __restrict__ wsP,
                                                float* __restrict__ outLoss) {
  __shared__ float sm[TPB];
  int t = threadIdx.x;
  float s = 0.f;
  for (int i = 0; i < 16; ++i) s += wsP[t + i * TPB];
  sm[t] = s;
  __syncthreads();
  if (t == 0) {
    float tot = 0.f;
    for (int i = 0; i < TPB; ++i) tot += sm[i];
    // loss = q_loss + 0.25*e_loss ; forward values equal => 1.25 * mean
    outLoss[0] = tot * (1.25f / (float)(NROWS * (size_t)DIMS));
  }
}

extern "C" void kernel_launch(void* const* d_in, const int* in_sizes, int n_in,
                              void* d_out, int out_size, void* d_ws, size_t ws_size,
                              hipStream_t stream) {
  const float* xin = (const float*)d_in[0];   // [B, D, T] f32
  const float* emb = (const float*)d_in[1];   // [K, D]    f32

  float* out     = (float*)d_out;
  float* outQ    = out;                                // 16777216 quantized
  float* outLoss = out + (size_t)BB * DIMS * TT;       // 1 loss
  float* outI    = outLoss + 1;                        // 262144 indices (as float)

  char* ws = (char*)d_ws;
  unsigned short* wsB  = (unsigned short*)ws;          // 64 KB swizzled bf16 codebook
  float*          wsEn = (float*)(ws + 65536);         // 2 KB |e|^2
  float*          wsP  = (float*)(ws + 65536 + 2048);  // 16 KB wave partials (4096)

  vq_prep <<<128, TPB, 0, stream>>>(emb, wsB, wsEn);
  vq_main <<<MAIN_BLOCKS, TPB, 0, stream>>>(xin, emb, wsB, wsEn, outQ, outI, wsP);
  vq_final<<<1, TPB, 0, stream>>>(wsP, outLoss);
}
